// TransRNet_49727131353819
// MI455X (gfx1250) — compile-verified, matrix operand order checked by reading, not run on hardware
//
#include <hip/hip_runtime.h>

typedef __attribute__((ext_vector_type(16))) _Float16 v16h;
typedef __attribute__((ext_vector_type(8)))  float    v8f;

#define DE 100
#define DR 100
#define RCNT 1000
#define BATCH 8192
#define NS 3
#define NTILES 7                    // N padded: 7*16 = 112 >= 100
#define KTILES 4                    // K padded: 4*32 = 128 >= 100
#define WAVES 8
#define TILE_HALFS 512              // 32 lanes * 16 halfs per WMMA B tile
#define REL_TILES (NTILES * KTILES) // 28 tiles per relation
#define REL_HALFS (REL_TILES * TILE_HALFS)

// K index inside a 32-wide K-tile for fragment element (j = e>>1, parity = e&1)
__device__ __forceinline__ int frag_kk(int j, int khalf) {
    return ((j >> 2) << 4) + 2 * (j & 3) + 8 * khalf;
}

// ---------------------------------------------------------------------------
// Pack relation_m (f32 [R][100][100]) into f16 WMMA-B-fragment layout:
//   ws[r][nt][kt][lane][e]  (contiguous 1KB per tile, 28KB per relation)
// ---------------------------------------------------------------------------
__global__ __launch_bounds__(256) void pack_relm_kernel(
    const float* __restrict__ relation_m, _Float16* __restrict__ wsm)
{
    const int r = blockIdx.x;
    const float* Mp = relation_m + (size_t)r * (DE * DR);
    _Float16* dst   = wsm + (size_t)r * REL_HALFS;

    for (int p = threadIdx.x; p < REL_HALFS / 2; p += 256) {
        const int idx0   = p * 2;            // even element index
        const int tile   = idx0 >> 9;
        const int within = idx0 & 511;
        const int lane   = within >> 4;
        const int e      = within & 15;      // even
        const int nt     = tile / KTILES;
        const int kt     = tile % KTILES;
        const int n      = nt * 16 + (lane & 15);
        const int khalf  = lane >> 4;
        const int j      = e >> 1;
        const int k      = kt * 32 + frag_kk(j, khalf);
        const bool nok   = (n < DR);
        const bool v0    = nok && (k < DE);
        const bool v1    = nok && (k + 1 < DE);
        const size_t o0  = v0 ? ((size_t)k * DR + n) : 0;
        const size_t o1  = v1 ? ((size_t)(k + 1) * DR + n) : 0;
        float f0 = Mp[o0]; f0 = v0 ? f0 : 0.f;
        float f1 = Mp[o1]; f1 = v1 ? f1 : 0.f;
        union { _Float16 h[2]; unsigned u; } pk;
        pk.h[0] = (_Float16)f0;
        pk.h[1] = (_Float16)f1;
        *(unsigned*)((char*)dst + (size_t)idx0 * 2) = pk.u;  // 4B-aligned
    }
}

// ---------------------------------------------------------------------------
// Main loss kernel: one wave per triplet; 8 entity rows (pos h/t + 3 neg h/t)
// projected through M_r with 7x4 v_wmma_f32_16x16x32_f16, B double-buffered.
// ---------------------------------------------------------------------------
template <bool PACKED>
__global__ __launch_bounds__(256) void transr_loss_kernel(
    const float* __restrict__ entity_emb,
    const float* __restrict__ relation_emb,
    const float* __restrict__ relation_m,
    const _Float16* __restrict__ wsm,
    const int* __restrict__ triplets,
    const int* __restrict__ neg,
    float* __restrict__ out)
{
    __shared__ __align__(32) _Float16 A_frag[WAVES][KTILES][32][16]; // 32 KB
    __shared__ float r_sh[WAVES][16 * NTILES];                       // 3.5 KB
    __shared__ float wsum[WAVES];

    const int tid   = threadIdx.x;
    const int wave  = tid >> 5;
    const int lane  = tid & 31;
    const int mrow  = lane & 15;
    const int khalf = lane >> 4;
    const int b     = blockIdx.x * WAVES + wave;

    // ---- indices (negatives keep the relation of the positive) ----
    const int h0 = triplets[3 * b + 0];
    const int r0 = triplets[3 * b + 1];
    const int t0 = triplets[3 * b + 2];
    int rows[8];
    rows[0] = h0;
    rows[1] = t0;
#pragma unroll
    for (int s = 0; s < NS; ++s) {
        rows[2 + 2 * s] = neg[(b * NS + s) * 3 + 0];
        rows[3 + 2 * s] = neg[(b * NS + s) * 3 + 2];
    }

    if (PACKED) {
        __builtin_prefetch(wsm + (size_t)r0 * REL_HALFS + lane * 448, 0, 1);
    } else {
        __builtin_prefetch(relation_m + (size_t)r0 * (DE * DR) + lane * 320, 0, 1);
    }

    // ---- zero-init this wave's A fragment slots (covers pad rows 8..15) ----
    const v16h zfrag = {};
#pragma unroll
    for (int kt = 0; kt < KTILES; ++kt)
        *(v16h*)&A_frag[wave][kt][lane][0] = zfrag;

    // ---- stage renormed entity rows directly in fragment layout ----
#pragma unroll
    for (int row = 0; row < 8; ++row) {
        const float* ep = entity_emb + (size_t)rows[row] * DE;
        float v[4];
        float ss = 0.f;
#pragma unroll
        for (int j = 0; j < 4; ++j) {
            const int k = lane + 32 * j;
            const bool ok = (k < DE);
            float raw = ep[ok ? k : 0];
            v[j] = ok ? raw : 0.f;
            ss += v[j] * v[j];
        }
#pragma unroll
        for (int m = 16; m >= 1; m >>= 1) ss += __shfl_xor(ss, m, 32);
        const float nrm   = sqrtf(ss);
        const float scale = (nrm > 1.0f) ? (1.0f / (nrm + 1e-7f)) : 1.0f;
#pragma unroll
        for (int j = 0; j < 4; ++j) {
            const int k      = lane + 32 * j;
            const int kt     = k >> 5;
            const int kk     = k & 31;
            const int parity = kk & 1;
            const int kh     = (kk >> 3) & 1;
            const int jf     = ((kk >> 4) & 1) * 4 + ((kk >> 1) & 3);
            A_frag[wave][kt][row + 16 * kh][2 * jf + parity] =
                (_Float16)(v[j] * scale);
        }
    }

    // ---- stage relation vector (padded to 112) ----
    const float* rp = relation_emb + (size_t)r0 * DR;
#pragma unroll
    for (int j = 0; j < 4; ++j) {
        const int k = lane + 32 * j;
        if (k < 16 * NTILES) {
            const bool ok = (k < DR);
            float raw = rp[ok ? k : 0];
            r_sh[wave][k] = ok ? raw : 0.f;
        }
    }

    __syncthreads();

    // ---- hoist A fragments and relation values into registers ----
    v16h afrag[KTILES];
#pragma unroll
    for (int kt = 0; kt < KTILES; ++kt)
        afrag[kt] = *(const v16h*)&A_frag[wave][kt][lane][0];

    float rv[NTILES];
#pragma unroll
    for (int nt = 0; nt < NTILES; ++nt)
        rv[nt] = r_sh[wave][nt * 16 + mrow];

    const float* Mp = relation_m + (size_t)r0 * (DE * DR);
    const _Float16* wrel = PACKED ? (wsm + (size_t)r0 * REL_HALFS) : nullptr;

    float sq0 = 0.f, sq1 = 0.f, sq2 = 0.f, sq3 = 0.f;
    const float lmask = (lane < 16) ? 1.0f : 0.0f;  // rows 0..7 live in lanes 0..15

    if (PACKED) {
        // ---- software-pipelined: double-buffer the 4 B fragments per N-tile ----
        v16h bbuf[2][KTILES];
#pragma unroll
        for (int kt = 0; kt < KTILES; ++kt)
            bbuf[0][kt] = *(const v16h*)(wrel + (size_t)kt * TILE_HALFS + lane * 16);

#pragma unroll
        for (int nt = 0; nt < NTILES; ++nt) {
            const int cur = nt & 1;
            const int nxt = cur ^ 1;
            if (nt + 1 < NTILES) {
#pragma unroll
                for (int kt = 0; kt < KTILES; ++kt)
                    bbuf[nxt][kt] = *(const v16h*)(
                        wrel + ((size_t)((nt + 1) * KTILES + kt)) * TILE_HALFS
                             + lane * 16);
            }
            v8f acc = {};
#pragma unroll
            for (int kt = 0; kt < KTILES; ++kt)
                acc = __builtin_amdgcn_wmma_f32_16x16x32_f16(
                    false, afrag[kt], false, bbuf[cur][kt], (short)0, acc,
                    false, false);

            // branch-free distance accumulation (masked FMA instead of exec)
            const float cmask = ((nt * 16 + mrow) < DR) ? lmask : 0.0f;
            float d0 = acc[0] + rv[nt] - acc[1] + 1e-6f; sq0 += cmask * d0 * d0;
            float d1 = acc[2] + rv[nt] - acc[3] + 1e-6f; sq1 += cmask * d1 * d1;
            float d2 = acc[4] + rv[nt] - acc[5] + 1e-6f; sq2 += cmask * d2 * d2;
            float d3 = acc[6] + rv[nt] - acc[7] + 1e-6f; sq3 += cmask * d3 * d3;
        }
    } else {
        // ---- fallback: gather B fragments from raw f32 M (branch-free) ----
        for (int nt = 0; nt < NTILES; ++nt) {
            const int  ncol = nt * 16 + mrow;
            const bool nok  = (ncol < DR);
            v8f acc = {};
#pragma unroll
            for (int kt = 0; kt < KTILES; ++kt) {
                v16h bfrag;
#pragma unroll
                for (int j = 0; j < 8; ++j) {
                    const int k0 = kt * 32 + frag_kk(j, khalf);
                    const bool v0 = nok && (k0 < DE);
                    const bool v1 = nok && (k0 + 1 < DE);
                    const size_t o0 = v0 ? ((size_t)k0 * DR + ncol) : 0;
                    const size_t o1 = v1 ? ((size_t)(k0 + 1) * DR + ncol) : 0;
                    float b0 = Mp[o0]; b0 = v0 ? b0 : 0.f;
                    float b1 = Mp[o1]; b1 = v1 ? b1 : 0.f;
                    bfrag[2 * j]     = (_Float16)b0;
                    bfrag[2 * j + 1] = (_Float16)b1;
                }
                acc = __builtin_amdgcn_wmma_f32_16x16x32_f16(
                    false, afrag[kt], false, bfrag, (short)0, acc, false, false);
            }
            const float cmask = (ncol < DR) ? lmask : 0.0f;
            float d0 = acc[0] + rv[nt] - acc[1] + 1e-6f; sq0 += cmask * d0 * d0;
            float d1 = acc[2] + rv[nt] - acc[3] + 1e-6f; sq1 += cmask * d1 * d1;
            float d2 = acc[4] + rv[nt] - acc[5] + 1e-6f; sq2 += cmask * d2 * d2;
            float d3 = acc[6] + rv[nt] - acc[7] + 1e-6f; sq3 += cmask * d3 * d3;
        }
    }

#pragma unroll
    for (int m = 16; m >= 1; m >>= 1) {
        sq0 += __shfl_xor(sq0, m, 32);
        sq1 += __shfl_xor(sq1, m, 32);
        sq2 += __shfl_xor(sq2, m, 32);
        sq3 += __shfl_xor(sq3, m, 32);
    }

    if (lane == 0) {
        const float pos = sqrtf(sq0);
        const float nm  = (sqrtf(sq1) + sqrtf(sq2) + sqrtf(sq3)) * (1.0f / 3.0f);
        const float v   = pos - nm + 1.0f;  // margin = 1.0
        wsum[wave] = (v > 0.f) ? (v * (1.0f / BATCH)) : 0.f;
    }
    __syncthreads();
    if (tid == 0) {
        float s = 0.f;
#pragma unroll
        for (int w = 0; w < WAVES; ++w) s += wsum[w];
        atomicAdd(out, s);
    }
}

extern "C" void kernel_launch(void* const* d_in, const int* in_sizes, int n_in,
                              void* d_out, int out_size, void* d_ws, size_t ws_size,
                              hipStream_t stream) {
    (void)in_sizes; (void)n_in; (void)out_size;
    const float* entity_emb   = (const float*)d_in[0];
    const float* relation_emb = (const float*)d_in[1];
    const float* relation_m   = (const float*)d_in[2];
    const int*   triplets     = (const int*)d_in[3];
    const int*   neg          = (const int*)d_in[4];
    float* out = (float*)d_out;

    hipMemsetAsync(out, 0, sizeof(float), stream);

    const size_t need = (size_t)RCNT * REL_HALFS * sizeof(_Float16); // ~27.3 MB
    dim3 grid(BATCH / WAVES);
    dim3 block(256);

    if (ws_size >= need) {
        _Float16* wsm = (_Float16*)d_ws;
        pack_relm_kernel<<<RCNT, 256, 0, stream>>>(relation_m, wsm);
        transr_loss_kernel<true><<<grid, block, 0, stream>>>(
            entity_emb, relation_emb, relation_m, wsm, triplets, neg, out);
    } else {
        transr_loss_kernel<false><<<grid, block, 0, stream>>>(
            entity_emb, relation_emb, relation_m, (const _Float16*)d_ws,
            triplets, neg, out);
    }
}
